// LADIES_Mini_Batch_ERGCN_7627861918261
// MI455X (gfx1250) — matrix operand order repaired and checked by reading, not compiled, not run on hardware
//
#include <hip/hip_runtime.h>

// Problem constants (match reference)
#define R_    16
#define NB_   16
#define N2_   8192
#define N1_   1024
#define NOUT_ 256
#define F_    128
#define E_    64
#define C_    32

typedef __attribute__((ext_vector_type(16))) __bf16 v16bf;
typedef __attribute__((ext_vector_type(8)))  float  v8f;

// ---------------------------------------------------------------------------
// WMMA helper: D = A(16x32 bf16) * B(32x16 bf16) + C(16x16 f32)
// ---------------------------------------------------------------------------
__device__ __forceinline__ v8f wmma_bf16(v16bf a, v16bf b, v8f c) {
    return __builtin_amdgcn_wmma_f32_16x16x32_bf16(
        /*neg_a=*/false, a, /*neg_b=*/false, b,
        /*c_mod=*/(short)0, c, /*reuse_a=*/false, /*reuse_b=*/false);
}

// A fragment (16-bit A 16x32 layout, ISA 7.12.2) from a row-major fp32 source.
// p must already point at (row, kbase + 8*(lane>>4)); 32B aligned.
__device__ __forceinline__ v16bf load_a_f32(const float* p) {
    float4 u0 = *(const float4*)(p + 0);
    float4 u1 = *(const float4*)(p + 4);
    float4 u2 = *(const float4*)(p + 16);
    float4 u3 = *(const float4*)(p + 20);
    v16bf a;
    a[0]=(__bf16)u0.x;  a[1]=(__bf16)u0.y;  a[2]=(__bf16)u0.z;  a[3]=(__bf16)u0.w;
    a[4]=(__bf16)u1.x;  a[5]=(__bf16)u1.y;  a[6]=(__bf16)u1.z;  a[7]=(__bf16)u1.w;
    a[8]=(__bf16)u2.x;  a[9]=(__bf16)u2.y;  a[10]=(__bf16)u2.z; a[11]=(__bf16)u2.w;
    a[12]=(__bf16)u3.x; a[13]=(__bf16)u3.y; a[14]=(__bf16)u3.z; a[15]=(__bf16)u3.w;
    return a;
}

// A fragment from a row-major bf16 source (two 16B chunks 32 elems apart in K
// would be wrong -- chunks are at +0 and +16 elements).
__device__ __forceinline__ v16bf load_a_bf16(const __bf16* p) {
    union { uint4 u[2]; v16bf v; } t;
    t.u[0] = *(const uint4*)(p);
    t.u[1] = *(const uint4*)(p + 16);
    return t.v;
}

// B fragment (32x16): lane holds one column, two 8-element K chunks at +0 and +16.
// Source stored K-major (BT[n][k]); p points at (col, kbase + 8*(lane>>4)).
__device__ __forceinline__ v16bf load_b_bf16(const __bf16* p) {
    union { uint4 u[2]; v16bf v; } t;
    const uint4* q = (const uint4*)p;
    t.u[0] = q[0];
    t.u[1] = *(const uint4*)(p + 16);
    return t.v;
}

// Pack 8 accumulator floats -> 8 contiguous bf16 (one 16B store).
__device__ __forceinline__ void store8_bf16(__bf16* p, v8f c) {
    union { __bf16 h[8]; uint4 u; } t;
#pragma unroll
    for (int i = 0; i < 8; ++i) t.h[i] = (__bf16)c[i];
    *(uint4*)p = t.u;
}

// ---------------------------------------------------------------------------
// k1: basis composition.
//   w1t[r][e][f] = sum_b comp1[r,b] * bases1[b,f,e]     (K-major for WMMA B)
//   w2t[r][c][h] = sum_b comp2[r,b] * bases2[b,h,c]
// ---------------------------------------------------------------------------
__global__ void k1_weights(const float* __restrict__ comp1,
                           const float* __restrict__ bases1,
                           const float* __restrict__ comp2,
                           const float* __restrict__ bases2,
                           __bf16* __restrict__ w1t,
                           __bf16* __restrict__ w2t) {
    const int idx = blockIdx.x * 256 + threadIdx.x;
    if (idx < R_ * E_ * F_) {
        const int r = idx / (E_ * F_);
        const int e = (idx / F_) % E_;
        const int f = idx % F_;
        float s = 0.f;
#pragma unroll
        for (int b = 0; b < NB_; ++b)
            s += comp1[r * NB_ + b] * bases1[((long)b * F_ + f) * E_ + e];
        w1t[idx] = (__bf16)s;                       // idx == (r*E_+e)*F_+f
    } else {
        const int j = idx - R_ * E_ * F_;
        if (j < R_ * C_ * E_) {
            const int r = j / (C_ * E_);
            const int c = (j / E_) % C_;
            const int h = j % E_;
            float s = 0.f;
#pragma unroll
            for (int b = 0; b < NB_; ++b)
                s += comp2[r * NB_ + b] * bases2[((long)b * E_ + h) * C_ + c];
            w2t[j] = (__bf16)s;                     // j == (r*C_+c)*E_+h
        }
    }
}

// ---------------------------------------------------------------------------
// k2: gather + per-relation feature transform.
//   xwT[e][r*N2 + n] = sum_f X[sel[n], f] * w1t[r][e][f]
// grid: (N2/16, R); block: 128 (4 waves, one E-tile of 16 each).
// ---------------------------------------------------------------------------
__global__ void k2_xw(const float* __restrict__ X,
                      const int*   __restrict__ sel,
                      const __bf16* __restrict__ w1t,
                      __bf16* __restrict__ xwT) {
    const int mt   = blockIdx.x;          // N2 tile
    const int r    = blockIdx.y;
    const int wave = threadIdx.x >> 5;    // E tile (0..3)
    const int lane = threadIdx.x & 31;
    const int row  = lane & 15;           // A row / B col
    const int half = lane >> 4;

    const int src = sel[mt * 16 + row];   // gather
    const float*  xrow = X + (long)src * F_;
    const __bf16* bt   = w1t + ((long)r * E_ + wave * 16 + row) * F_;

    v8f acc = {};
#pragma unroll
    for (int kk = 0; kk < F_ / 32; ++kk) {
        const int kbase = kk * 32;
        v16bf a = load_a_f32(xrow + kbase + half * 8);
        v16bf b = load_b_bf16(bt + kbase + half * 8);
        acc = wmma_bf16(a, b, acc);
    }
    // store D transposed: (row m_local = i + 8*half, col = lane&15)
    const long K = (long)R_ * N2_;
    __bf16* dst = xwT + (long)(wave * 16 + row) * K
                      + (long)r * N2_ + mt * 16 + half * 8;
    store8_bf16(dst, acc);
}

// ---------------------------------------------------------------------------
// k3: the HBM-bound matmul.  h1 = relu(A0 @ xw + bias1), stored bf16 [N1][E].
// grid: N1/16 blocks; block 256 = 8 waves, each wave owns a K/8 slice and all
// four E-tiles (A0 streamed exactly once).  LDS cross-wave reduction.
// ---------------------------------------------------------------------------
__global__ void k3_h1(const float* __restrict__ A0,
                      const __bf16* __restrict__ xwT,
                      const float* __restrict__ bias1,
                      __bf16* __restrict__ h1) {
    __shared__ float red[8 * 1024];       // 8 waves x 16x64 tile = 32 KB
    const int mt   = blockIdx.x;          // 0..63
    const int wave = threadIdx.x >> 5;
    const int lane = threadIdx.x & 31;
    const int row  = lane & 15;
    const int half = lane >> 4;

    const long K  = (long)R_ * N2_;       // 131072
    const long k0 = (long)wave * (K / 8);
    const float* arow = A0 + (long)(mt * 16 + row) * K;

    v8f acc[4] = {};
    const int steps = (int)(K / 8 / 32);  // 512
    for (int s = 0; s < steps; ++s) {
        const long kbase = k0 + (long)s * 32;
        // prefetch the A0 stream ~2KB ahead (global_prefetch_b8)
        __builtin_prefetch((const void*)(arow + kbase + 512), 0, 0);
        v16bf a = load_a_f32(arow + kbase + half * 8);
#pragma unroll
        for (int et = 0; et < 4; ++et) {
            v16bf b = load_b_bf16(xwT + (long)(et * 16 + row) * K
                                      + kbase + half * 8);
            acc[et] = wmma_bf16(a, b, acc[et]);
        }
    }
#pragma unroll
    for (int et = 0; et < 4; ++et)
#pragma unroll
        for (int i = 0; i < 8; ++i)
            red[wave * 1024 + et * 256 + i * 32 + lane] = acc[et][i];
    __syncthreads();

    // 1024 outputs / 256 threads = 4 each: reduce 8 waves, bias, relu, store.
    for (int j = threadIdx.x; j < 1024; j += 256) {
        const int mloc = j >> 6;          // 0..15
        const int e    = j & 63;
        const int et   = e >> 4;
        const int ln   = (e & 15) + ((mloc >> 3) << 4);
        const int i2   = mloc & 7;
        float s = 0.f;
#pragma unroll
        for (int w = 0; w < 8; ++w)
            s += red[w * 1024 + et * 256 + i2 * 32 + ln];
        s += bias1[e];
        s = s > 0.f ? s : 0.f;
        h1[(mt * 16 + mloc) * E_ + e] = (__bf16)s;
    }
}

// ---------------------------------------------------------------------------
// k4a: h2T[c][r*N1 + n] = sum_h h1[n,h] * w2t[r][c][h]
// grid: (N1/16, R); block 64 = 2 waves (one C-tile each); K = 64 (2 steps).
// ---------------------------------------------------------------------------
__global__ void k4a_h2(const __bf16* __restrict__ h1,
                       const __bf16* __restrict__ w2t,
                       __bf16* __restrict__ h2T) {
    const int mt   = blockIdx.x;
    const int r    = blockIdx.y;
    const int wave = threadIdx.x >> 5;    // C tile (0..1)
    const int lane = threadIdx.x & 31;
    const int row  = lane & 15;
    const int half = lane >> 4;

    const __bf16* arow = h1 + (long)(mt * 16 + row) * E_;
    const __bf16* bt   = w2t + ((long)r * C_ + wave * 16 + row) * E_;

    v8f acc = {};
#pragma unroll
    for (int kk = 0; kk < E_ / 32; ++kk) {
        const int kbase = kk * 32;
        v16bf a = load_a_bf16(arow + kbase + half * 8);
        v16bf b = load_b_bf16(bt + kbase + half * 8);
        acc = wmma_bf16(a, b, acc);
    }
    const long K = (long)R_ * N1_;        // 16384
    __bf16* dst = h2T + (long)(wave * 16 + row) * K
                      + (long)r * N1_ + mt * 16 + half * 8;
    store8_bf16(dst, acc);
}

// ---------------------------------------------------------------------------
// k4b: out = A1 @ h2 + bias2  (fp32 out [NOUT][C]).
// grid: NOUT/16 blocks; block 256 = 8 waves splitting K=16384; LDS reduce.
// ---------------------------------------------------------------------------
__global__ void k4b_out(const float* __restrict__ A1,
                        const __bf16* __restrict__ h2T,
                        const float* __restrict__ bias2,
                        float* __restrict__ out) {
    __shared__ float red[8 * 512];        // 16 KB
    const int mt   = blockIdx.x;          // 0..15
    const int wave = threadIdx.x >> 5;
    const int lane = threadIdx.x & 31;
    const int row  = lane & 15;
    const int half = lane >> 4;

    const long K  = (long)R_ * N1_;       // 16384
    const long k0 = (long)wave * (K / 8); // 2048
    const float* arow = A1 + (long)(mt * 16 + row) * K;

    v8f acc[2] = {};
    for (int s = 0; s < (int)(K / 8 / 32); ++s) {   // 64 steps
        const long kbase = k0 + (long)s * 32;
        v16bf a = load_a_f32(arow + kbase + half * 8);
#pragma unroll
        for (int ct = 0; ct < 2; ++ct) {
            v16bf b = load_b_bf16(h2T + (long)(ct * 16 + row) * K
                                      + kbase + half * 8);
            acc[ct] = wmma_bf16(a, b, acc[ct]);
        }
    }
#pragma unroll
    for (int ct = 0; ct < 2; ++ct)
#pragma unroll
        for (int i = 0; i < 8; ++i)
            red[wave * 512 + ct * 256 + i * 32 + lane] = acc[ct][i];
    __syncthreads();

    for (int j = threadIdx.x; j < 512; j += 256) {
        const int mloc = j >> 5;          // 0..15
        const int c    = j & 31;
        const int ct   = c >> 4;
        const int ln   = (c & 15) + ((mloc >> 3) << 4);
        const int i2   = mloc & 7;
        float s = 0.f;
#pragma unroll
        for (int w = 0; w < 8; ++w)
            s += red[w * 512 + ct * 256 + i2 * 32 + ln];
        out[(mt * 16 + mloc) * C_ + c] = s + bias2[c];
    }
}

// ---------------------------------------------------------------------------
// launch
// ---------------------------------------------------------------------------
extern "C" void kernel_launch(void* const* d_in, const int* in_sizes, int n_in,
                              void* d_out, int out_size, void* d_ws, size_t ws_size,
                              hipStream_t stream) {
    const float* X      = (const float*)d_in[0];
    const int*   sel    = (const int*)  d_in[1];
    const float* A0     = (const float*)d_in[2];
    const float* A1     = (const float*)d_in[3];
    const float* comp1  = (const float*)d_in[4];
    const float* bases1 = (const float*)d_in[5];
    const float* comp2  = (const float*)d_in[6];
    const float* bases2 = (const float*)d_in[7];
    const float* bias1  = (const float*)d_in[8];
    const float* bias2  = (const float*)d_in[9];
    float* out = (float*)d_out;

    // workspace carve-up (bf16 intermediates), 256B-aligned offsets
    char* ws = (char*)d_ws;
    __bf16* w1t = (__bf16*)(ws);                       // 16*64*128*2  = 256 KB
    __bf16* w2t = (__bf16*)(ws + (256u << 10));        // 16*32*64*2   =  64 KB
    __bf16* xwT = (__bf16*)(ws + (320u << 10));        // 64*131072*2  =  16 MB
    __bf16* h1  = (__bf16*)(ws + (320u << 10) + (16u << 20));          // 128 KB
    __bf16* h2T = (__bf16*)(ws + (320u << 10) + (16u << 20) + (128u << 10)); // 1 MB

    // k1: compose relation weights into bf16, K-major for WMMA-B use
    k1_weights<<<(R_*E_*F_ + R_*C_*E_ + 255) / 256, 256, 0, stream>>>(
        comp1, bases1, comp2, bases2, w1t, w2t);

    // k2: gather + x @ w1[r]  -> xwT (bf16, K-major, L2-resident 16 MB)
    k2_xw<<<dim3(N2_ / 16, R_), 128, 0, stream>>>(X, sel, w1t, xwT);

    // k3: h1 = relu(A0 @ xw + bias1)  (A0 streamed once: HBM-bound)
    k3_h1<<<N1_ / 16, 256, 0, stream>>>(A0, xwT, bias1, h1);

    // k4a: h2T = (h1 @ w2[r]) transposed bf16
    k4a_h2<<<dim3(N1_ / 16, R_), 64, 0, stream>>>(h1, w2t, h2T);

    // k4b: out = A1 @ h2 + bias2
    k4b_out<<<NOUT_ / 16, 256, 0, stream>>>(A1, h2T, bias2, out);
}
